// Model_47957604827446
// MI455X (gfx1250) — compile-verified, matrix operand order checked by reading, not run on hardware
//
#include <hip/hip_runtime.h>
#include <hip/hip_bf16.h>

typedef __attribute__((ext_vector_type(16))) _Float16 v16h;
typedef __attribute__((ext_vector_type(8)))  float    v8f;

#define B_TOK 8192
#define NEXP  8

// LDS tile row stride in halves: 88 halves = 176 B = 44 banks.
// gcd(44,64)=4 and 44*r mod 64 is distinct for r=0..15 -> conflict-free
// operand fetch; 176 is a multiple of 16 B so b128 LDS ops stay aligned.
#define LDSTR  88
#define TILESZ (64 * LDSTR)   // halves per 64x64 tile

// ---------------------------------------------------------------------------
// Gating MLP: g = ((p@Wg1^T+bg1)@Wg2^T+bg2)@Wg3^T+bg3   [B,8]
// ---------------------------------------------------------------------------
__global__ __launch_bounds__(256) void gating_kernel(
    const float* __restrict__ p,
    const float* __restrict__ Wg1, const float* __restrict__ bg1,
    const float* __restrict__ Wg2, const float* __restrict__ bg2,
    const float* __restrict__ Wg3, const float* __restrict__ bg3,
    float* __restrict__ g)
{
    int b = blockIdx.x * blockDim.x + threadIdx.x;
    if (b >= B_TOK) return;
    float pv[12];
    #pragma unroll
    for (int i = 0; i < 12; ++i) pv[i] = p[b * 12 + i];
    float h1[32];
    #pragma unroll
    for (int j = 0; j < 32; ++j) {
        float s = bg1[j];
        #pragma unroll
        for (int i = 0; i < 12; ++i) s += Wg1[j * 12 + i] * pv[i];
        h1[j] = s;
    }
    float h2[32];
    #pragma unroll
    for (int j = 0; j < 32; ++j) {
        float s = bg2[j];
        #pragma unroll
        for (int i = 0; i < 32; ++i) s += Wg2[j * 32 + i] * h1[i];
        h2[j] = s;
    }
    #pragma unroll
    for (int e = 0; e < 8; ++e) {
        float s = bg3[e];
        #pragma unroll
        for (int i = 0; i < 32; ++i) s += Wg3[e * 32 + i] * h2[i];
        g[b * 8 + e] = s;
    }
}

// ---------------------------------------------------------------------------
// fp32 -> f16 conversion (vectorized, grid-stride), n must be multiple of 4
// ---------------------------------------------------------------------------
__global__ __launch_bounds__(256) void cvt_f32_to_f16_kernel(
    const float* __restrict__ src, _Float16* __restrict__ dst, long long n4)
{
    long long i = (long long)blockIdx.x * blockDim.x + threadIdx.x;
    long long stride = (long long)gridDim.x * blockDim.x;
    for (; i < n4; i += stride) {
        float4 v = ((const float4*)src)[i];
        alignas(8) _Float16 t[4];
        t[0] = (_Float16)v.x; t[1] = (_Float16)v.y;
        t[2] = (_Float16)v.z; t[3] = (_Float16)v.w;
        *(float2*)(dst + i * 4) = *(float2*)t;
    }
}

// ---------------------------------------------------------------------------
// One MoE layer: out[b,o] = act( sum_e g[b,e] * ( (A @ W_e^T)[b,o] + bias_e[o] ) )
//   A    : [B, Din]  f16
//   W    : [E, Dout, Din] f32 (converted to f16 while staging to LDS)
//   bias : [E, Dout] f32, g : [B, 8] f32
// Tile: 64x64 per 256-thread block (8 waves, 4M x 2N, each wave 16x32).
// 8 expert accumulators live in registers; gates folded in the epilogue.
// ---------------------------------------------------------------------------
template <bool APPLY_ELU, bool STORE_F16>
__global__ __launch_bounds__(256) void moe_layer_kernel(
    const _Float16* __restrict__ Ah,
    const float*    __restrict__ W,
    const float*    __restrict__ bias,
    const float*    __restrict__ g,
    _Float16* __restrict__ outH,
    float*    __restrict__ outF,
    int Din, int Dout)
{
    // LDS: A tile 64x64 (stride 88 halves, conflict-free, 16B-aligned rows),
    // 4 expert B tiles, gate tile [64][8] f32, bias tile [8][64] f32. ~59 KB.
    __shared__ _Float16 sA[TILESZ];
    __shared__ _Float16 sB[4 * TILESZ];
    __shared__ float    sG[64 * 8];
    __shared__ float    sBias[8 * 64];

    const int tid  = threadIdx.x;
    const int w    = tid >> 5;          // wave 0..7
    const int wm   = w & 3;             // M sub-tile 0..3
    const int wn   = w >> 2;            // N half 0..1
    const int lane = tid & 31;
    const int lh   = lane >> 4;         // lane half (K-chunk select / M-row half)
    const int lm   = lane & 15;         // row/col within 16-tile

    const int tileM = blockIdx.y * 64;
    const int tileN = blockIdx.x * 64;

    // stage gates (contiguous: g is [B][8]) and bias tile
    for (int i = tid; i < 512; i += 256) sG[i] = g[(size_t)tileM * 8 + i];
    for (int i = tid; i < 512; i += 256) {
        int e = i >> 6, j = i & 63;
        sBias[i] = bias[(size_t)e * Dout + tileN + j];
    }

    v8f acc[NEXP][2];
    {
        v8f z = {};
        #pragma unroll
        for (int e = 0; e < NEXP; ++e)
            #pragma unroll
            for (int t = 0; t < 2; ++t) acc[e][t] = z;
    }

    const int aoff = (16 * wm + lm) * LDSTR;

    for (int k0 = 0; k0 < Din; k0 += 64) {
        #pragma unroll
        for (int p = 0; p < 2; ++p) {        // two groups of 4 experts
            __syncthreads();                 // protect prior LDS reads

            if (p == 0) {
                // stage A tile: 64 rows x 64 halves (f16 source, b128 copies)
                for (int u = tid; u < 512; u += 256) {
                    int row = u >> 3, ch = u & 7;
                    float4 v = *(const float4*)(Ah + (size_t)(tileM + row) * Din + k0 + ch * 8);
                    *(float4*)(&sA[row * LDSTR + ch * 8]) = v;
                }
            }
            // stage 4 expert B tiles, converting fp32 -> f16 on the fly
            for (int u = tid; u < 2048; u += 256) {
                int el  = u >> 9;            // expert-local 0..3
                int row = (u >> 3) & 63;
                int ch  = u & 7;
                const float* wp = W + ((size_t)(p * 4 + el) * Dout + tileN + row) * Din
                                    + k0 + ch * 8;
                float4 w0 = *(const float4*)(wp);
                float4 w1 = *(const float4*)(wp + 4);
                alignas(16) _Float16 t8[8];
                t8[0] = (_Float16)w0.x; t8[1] = (_Float16)w0.y;
                t8[2] = (_Float16)w0.z; t8[3] = (_Float16)w0.w;
                t8[4] = (_Float16)w1.x; t8[5] = (_Float16)w1.y;
                t8[6] = (_Float16)w1.z; t8[7] = (_Float16)w1.w;
                *(float4*)(&sB[el * TILESZ + row * LDSTR + ch * 8]) = *(float4*)t8;
            }
            __syncthreads();

            // A operands for both K=32 steps (ISA 16-bit A layout:
            // lane = (half,m); halves = A[m, 8h..8h+7] ++ A[m, 16+8h..16+8h+7])
            v16h aR[2];
            #pragma unroll
            for (int kkI = 0; kkI < 2; ++kkI) {
                int ko = kkI * 32 + 8 * lh;
                float4 lo = *(const float4*)(&sA[aoff + ko]);
                float4 hi = *(const float4*)(&sA[aoff + ko + 16]);
                v16h a;
                *(float4*)&a         = lo;
                *(((float4*)&a) + 1) = hi;
                aR[kkI] = a;
            }

            #pragma unroll
            for (int el = 0; el < 4; ++el) {
                const int e = p * 4 + el;
                #pragma unroll
                for (int t = 0; t < 2; ++t) {
                    const int boff = el * TILESZ + (32 * wn + 16 * t + lm) * LDSTR;
                    #pragma unroll
                    for (int kkI = 0; kkI < 2; ++kkI) {
                        int ko = kkI * 32 + 8 * lh;
                        float4 lo = *(const float4*)(&sB[boff + ko]);
                        float4 hi = *(const float4*)(&sB[boff + ko + 16]);
                        v16h b;
                        *(float4*)&b         = lo;
                        *(((float4*)&b) + 1) = hi;
                        acc[e][t] = __builtin_amdgcn_wmma_f32_16x16x32_f16(
                            false, aR[kkI], false, b, (short)0, acc[e][t],
                            false, false);
                    }
                }
            }
        }
    }

    // ---- epilogue: gated combine + bias, ELU, store -------------------------
    float bias8[2][8];
    #pragma unroll
    for (int t = 0; t < 2; ++t)
        #pragma unroll
        for (int e = 0; e < 8; ++e)
            bias8[t][e] = sBias[e * 64 + 32 * wn + 16 * t + lm];

    #pragma unroll
    for (int r = 0; r < 8; ++r) {
        const int rowLocal = 16 * wm + 8 * lh + r;   // C/D layout: M = r + 8*half
        const int rowGlob  = tileM + rowLocal;
        float4 g0 = *(const float4*)(&sG[rowLocal * 8]);
        float4 g1 = *(const float4*)(&sG[rowLocal * 8 + 4]);
        float g8[8] = {g0.x, g0.y, g0.z, g0.w, g1.x, g1.y, g1.z, g1.w};
        #pragma unroll
        for (int t = 0; t < 2; ++t) {
            const int colLocal = 32 * wn + 16 * t + lm;
            float res = 0.f;
            #pragma unroll
            for (int e = 0; e < 8; ++e)
                res += g8[e] * (acc[e][t][r] + bias8[t][e]);
            if (APPLY_ELU) res = res > 0.f ? res : (__expf(res) - 1.f);
            size_t oidx = (size_t)rowGlob * Dout + tileN + colLocal;
            if (STORE_F16) outH[oidx] = (_Float16)res;
            else           outF[oidx] = res;
        }
    }
}

// ---------------------------------------------------------------------------
extern "C" void kernel_launch(void* const* d_in, const int* in_sizes, int n_in,
                              void* d_out, int out_size, void* d_ws, size_t ws_size,
                              hipStream_t stream)
{
    (void)in_sizes; (void)n_in; (void)out_size; (void)ws_size;

    const float* x   = (const float*)d_in[0];
    const float* p   = (const float*)d_in[1];
    // d_in[2] = z (unused by reference)
    const float* W0  = (const float*)d_in[3];
    const float* b0  = (const float*)d_in[4];
    const float* W1  = (const float*)d_in[5];
    const float* b1  = (const float*)d_in[6];
    const float* W2  = (const float*)d_in[7];
    const float* b2  = (const float*)d_in[8];
    const float* Wg1 = (const float*)d_in[9];
    const float* bg1 = (const float*)d_in[10];
    const float* Wg2 = (const float*)d_in[11];
    const float* bg2 = (const float*)d_in[12];
    const float* Wg3 = (const float*)d_in[13];
    const float* bg3 = (const float*)d_in[14];
    float* out = (float*)d_out;

    // workspace layout (~75 MB)
    char* ws = (char*)d_ws;
    float* gW = (float*)ws;                 ws += (size_t)B_TOK * 8 * sizeof(float);
    _Float16* xh = (_Float16*)ws;           ws += (size_t)B_TOK * 512  * sizeof(_Float16);
    _Float16* h1 = (_Float16*)ws;           ws += (size_t)B_TOK * 2048 * sizeof(_Float16);
    _Float16* h2 = (_Float16*)ws;           ws += (size_t)B_TOK * 2048 * sizeof(_Float16);

    // 1) gating
    gating_kernel<<<B_TOK / 256, 256, 0, stream>>>(p, Wg1, bg1, Wg2, bg2, Wg3, bg3, gW);

    // 2) x -> f16
    cvt_f32_to_f16_kernel<<<2048, 256, 0, stream>>>(x, xh, (long long)B_TOK * 512 / 4);

    // 3) layer 0: [B,512] -> [B,2048], ELU, f16 out
    moe_layer_kernel<true, true><<<dim3(2048 / 64, B_TOK / 64), 256, 0, stream>>>(
        xh, W0, b0, gW, h1, nullptr, 512, 2048);

    // 4) layer 1: [B,2048] -> [B,2048], ELU, f16 out
    moe_layer_kernel<true, true><<<dim3(2048 / 64, B_TOK / 64), 256, 0, stream>>>(
        h1, W1, b1, gW, h2, nullptr, 2048, 2048);

    // 5) layer 2: [B,2048] -> [B,512], fp32 out (no activation)
    moe_layer_kernel<false, false><<<dim3(512 / 64, B_TOK / 64), 256, 0, stream>>>(
        h2, W2, b2, gW, nullptr, out, 2048, 512);
}
